// AttentionBlock_86526411145692
// MI455X (gfx1250) — matrix operand order repaired and checked by reading, not compiled, not run on hardware
//
#include <hip/hip_runtime.h>

// ---------------------------------------------------------------------------
// AttentionBlock for MI455X (gfx1250, wave32, WMMA 16x16x32 f16->f32).
// Compute-bound (~35 GFLOP on ~16 MiB of activations, all L2-resident at
// 192 MB / 23.3 TB/s), so every matmul stage runs on v_wmma_f32_16x16x32_f16.
// GEMM tiles are double-buffered and staged with the CDNA5 async path
// (global_load_async_to_lds_b128 + s_wait_asynccnt) to overlap HBM/L2 fetch
// with the WMMA pipe.
// ---------------------------------------------------------------------------

#define DEV __device__ __forceinline__

typedef __attribute__((ext_vector_type(16))) _Float16 f16x16;
typedef __attribute__((ext_vector_type(8)))  _Float16 f16x8;
typedef __attribute__((ext_vector_type(8)))  float    f32x8;

namespace cfg {
constexpr int B      = 8;
constexpr int C      = 512;
constexpr int H      = 32;
constexpr int W      = 32;
constexpr int NTOK   = H * W;          // 1024 tokens
constexpr int HEADS  = 8;
constexpr int HD     = C / HEADS;      // 64
constexpr int GROUPS = 32;
constexpr int CPG    = C / GROUPS;     // 16
constexpr float EPS  = 1e-5f;
constexpr float QSCALE = 0.125f;       // 64^-0.5
}

// ---- fragment load helpers -------------------------------------------------
DEV f16x16 ld16(const _Float16* p) {
  f16x8 lo = *(const f16x8*)p;
  f16x8 hi = *(const f16x8*)(p + 8);
  f16x16 r;
#pragma unroll
  for (int i = 0; i < 8; ++i) { r[i] = lo[i]; r[i + 8] = hi[i]; }
  return r;
}

// A-matrix 16x32 f16 fragment from a row-major row pointer.
// lane m = lane&15, element j -> K = j + 16*half.
DEV f16x16 ld_afrag(const _Float16* row, int hf) {
  f16x8 lo = *(const f16x8*)(row + 16 * hf);
  f16x8 hi = *(const f16x8*)(row + 16 * hf + 8);
  f16x16 r;
#pragma unroll
  for (int i = 0; i < 8; ++i) { r[i] = lo[i]; r[i + 8] = hi[i]; }
  return r;
}

// B-matrix 32x16 fragment from N-major (transposed) storage Bt[n][k]:
// lane n = lane&15, element j -> K = j + 16*half  => 16 contiguous halves.
DEV f16x16 ld_bfrag(const _Float16* bt_row, int hf) {
  return ld16(bt_row + 16 * hf);
}

DEV f32x8 wmma(f16x16 a, f16x16 b, f32x8 c) {
  return __builtin_amdgcn_wmma_f32_16x16x32_f16(false, a, false, b,
                                                (short)0, c, false, false);
}

// CDNA5 async global->LDS b128 (16 bytes per lane, ASYNCcnt-tracked).
// Generic LDS pointer low 32 bits are the LDS byte offset (ISA 10.2).
DEV void async_b128(void* lds_dst, const void* gsrc) {
  unsigned int       l = (unsigned int)(unsigned long long)lds_dst;
  unsigned long long g = (unsigned long long)gsrc;
  asm volatile("global_load_async_to_lds_b128 %0, %1, off"
               :: "v"(l), "v"(g) : "memory");
}
DEV void wait_async0() { asm volatile("s_wait_asynccnt 0" ::: "memory"); }

// ---------------------------------------------------------------------------
// Kernel 1: GroupNorm, fp32 in -> f16 normalized activations [B][C][NTOK]
// ---------------------------------------------------------------------------
__global__ __launch_bounds__(256) void gn_kernel(
    const float* __restrict__ x, const float* __restrict__ gam,
    const float* __restrict__ bet, _Float16* __restrict__ xn) {
  using namespace cfg;
  __shared__ float s1[256], s2[256];
  const int b = blockIdx.x / GROUPS, g = blockIdx.x % GROUPS;
  const int elems = CPG * NTOK;  // 16384
  const size_t base = (size_t)b * C * NTOK + (size_t)g * CPG * NTOK;
  float sum = 0.f, ssq = 0.f;
  for (int e = threadIdx.x; e < elems; e += 256) {
    float v = x[base + e];
    sum += v; ssq += v * v;
  }
  s1[threadIdx.x] = sum; s2[threadIdx.x] = ssq;
  __syncthreads();
  for (int s = 128; s > 0; s >>= 1) {
    if ((int)threadIdx.x < s) {
      s1[threadIdx.x] += s1[threadIdx.x + s];
      s2[threadIdx.x] += s2[threadIdx.x + s];
    }
    __syncthreads();
  }
  const float mean = s1[0] / elems;
  const float var  = s2[0] / elems - mean * mean;
  const float inv  = rsqrtf(var + EPS);
  for (int e = threadIdx.x; e < elems; e += 256) {
    const int c = g * CPG + (e >> 10);
    const float v = (x[base + e] - mean) * inv * gam[c] + bet[c];
    xn[base + e] = (_Float16)v;
  }
}

// ---------------------------------------------------------------------------
// Kernel 2: weights fp32 -> f16 (done once, L2-resident afterwards)
// ---------------------------------------------------------------------------
__global__ __launch_bounds__(256) void cvt_kernel(
    const float* __restrict__ qw, const float* __restrict__ pw,
    _Float16* __restrict__ qw16, _Float16* __restrict__ pw16) {
  using namespace cfg;
  const int i = blockIdx.x * 256 + threadIdx.x;
  if (i < 3 * C * C) qw16[i] = (_Float16)qw[i];
  if (i < C * C)     pw16[i] = (_Float16)pw[i];
}

// ---------------------------------------------------------------------------
// Kernel 3/5: tiled WMMA GEMM  out[M,N] = W[M,K] * X[K,N]  (per batch z).
// Block tile 128x128, 8 waves (2x4), wave tile 64x32, K-step 32.
// Double-buffered LDS; A tile staged via async global->LDS (no VGPR
// round-trip), B tile global loads hoisted above the WMMA block.
// mode 0: QKV epilogue (bias, q-scale, head-major f16 q/k/v scatter)
// mode 1: proj epilogue (bias + residual, fp32 out)
// ---------------------------------------------------------------------------
__global__ __launch_bounds__(256) void gemm_kernel(
    const _Float16* __restrict__ Wm, const _Float16* __restrict__ Xall,
    const float* __restrict__ bias,
    _Float16* __restrict__ qout, _Float16* __restrict__ kout,
    _Float16* __restrict__ vout,
    const float* __restrict__ resid, float* __restrict__ fout, int mode) {
  using namespace cfg;
  constexpr int K = C;          // 512 for both GEMMs
  constexpr int BK = 32;
  constexpr int LDA = BK + 8;   // 40 halves (16B-multiple row stride)
  __shared__ _Float16 As[2][128][LDA];   // A tiles, row-major [m][k]  20 KB
  __shared__ _Float16 Bt[2][128][LDA];   // B tiles, transposed [n][k] 20 KB

  const int b  = blockIdx.z;
  const int n0 = blockIdx.x * 128;
  const int m0 = blockIdx.y * 128;
  const _Float16* X = Xall + (size_t)b * K * NTOK;

  const int tid = threadIdx.x;
  const int wid = tid >> 5, lane = tid & 31;
  const int wm = wid >> 2, wn = wid & 3;        // 2x4 wave grid
  const int hf = lane >> 4, l15 = lane & 15;

  // staging roles
  const int rowA = tid >> 1, segA = (tid & 1) * 16;   // A: 16 halves/thread
  const int kkB  = tid >> 3, nsB  = (tid & 7) * 16;   // B: 16 halves/thread
  const _Float16* Abase = Wm + (size_t)(m0 + rowA) * K + segA;
  const _Float16* Bbase = X + (size_t)kkB * NTOK + n0 + nsB;

  auto stageA = [&](int buf, int k0) {
    _Float16* d = &As[buf][rowA][segA];
    const _Float16* s = Abase + k0;
    async_b128(d, s);
    async_b128(d + 8, s + 8);
  };
  auto loadB = [&](int k0, f16x8& x0, f16x8& x1) {
    const _Float16* src = Bbase + (size_t)k0 * NTOK;
    x0 = *(const f16x8*)src;
    x1 = *(const f16x8*)(src + 8);
  };
  auto scatterB = [&](int buf, f16x8 x0, f16x8 x1) {
#pragma unroll
    for (int j = 0; j < 8; ++j) {
      Bt[buf][nsB + j][kkB]     = x0[j];
      Bt[buf][nsB + 8 + j][kkB] = x1[j];
    }
  };

  // prologue: stage tile 0
  stageA(0, 0);
  {
    f16x8 x0, x1;
    loadB(0, x0, x1);
    scatterB(0, x0, x1);
  }
  wait_async0();
  __syncthreads();

  f32x8 acc[4][2] = {};

  for (int k0 = 0; k0 < K; k0 += BK) {
    const int cur = (k0 >> 5) & 1, nxt = cur ^ 1;
    const bool more = (k0 + BK) < K;

    f16x8 x0, x1;
    if (more) {                 // overlap next-tile fetch with WMMA block
      stageA(nxt, k0 + BK);
      loadB(k0 + BK, x0, x1);
    }

    f16x16 bf[2];
#pragma unroll
    for (int fn = 0; fn < 2; ++fn)
      bf[fn] = ld_bfrag(&Bt[cur][wn * 32 + fn * 16 + l15][0], hf);
#pragma unroll
    for (int fm = 0; fm < 4; ++fm) {
      f16x16 af = ld_afrag(&As[cur][wm * 64 + fm * 16 + l15][0], hf);
#pragma unroll
      for (int fn = 0; fn < 2; ++fn)
        acc[fm][fn] = wmma(af, bf[fn], acc[fm][fn]);
    }

    if (more) scatterB(nxt, x0, x1);
    wait_async0();
    __syncthreads();
  }

  // epilogue: C/D layout -> row m = r + 8*half, col n = lane&15
#pragma unroll
  for (int fm = 0; fm < 4; ++fm) {
#pragma unroll
    for (int fn = 0; fn < 2; ++fn) {
      const int ng = n0 + wn * 32 + fn * 16 + l15;
#pragma unroll
      for (int r = 0; r < 8; ++r) {
        const int o = m0 + wm * 64 + fm * 16 + r + 8 * hf;
        float v = acc[fm][fn][r] + bias[o];
        if (mode == 0) {
          const int part = o >> 9, cc = o & 511;      // C == 512
          _Float16* dst = (part == 0) ? qout : (part == 1) ? kout : vout;
          if (part == 0) v *= QSCALE;
          dst[(size_t)b * C * NTOK + (size_t)cc * NTOK + ng] = (_Float16)v;
        } else {
          const size_t idx = (size_t)b * C * NTOK + (size_t)o * NTOK + ng;
          fout[idx] = v + resid[idx];
        }
      }
    }
  }
}

// ---------------------------------------------------------------------------
// Kernel 4: flash attention. Block = 128 queries of one (b,head).
// q,k,v layouts: [B*HEADS][HD=64][NTOK=1024] f16 (q pre-scaled).
// Each wave owns 16 query rows; S via WMMA, online softmax, P routed through
// per-wave LDS back to A-fragments, P*V^T with B-fragments straight from
// global (V's [d][j] layout is already Bt-major, 32B aligned).
// ---------------------------------------------------------------------------
__global__ __launch_bounds__(256) void attn_kernel(
    const _Float16* __restrict__ Q, const _Float16* __restrict__ Kg,
    const _Float16* __restrict__ V, _Float16* __restrict__ O) {
  using namespace cfg;
  constexpr int LQ = HD + 8;    // 72 halves
  constexpr int LP = 128 + 8;   // 136 halves
  __shared__ _Float16 Qs[128][LQ];       // [i][d]   (transposed)  18 KB
  __shared__ _Float16 Ks[128][LQ];       // [j][d]   (transposed)  18 KB
  __shared__ _Float16 Ps[8][16][LP];     // per-wave P tile        34 KB

  const int bh = blockIdx.y;
  const int i0 = blockIdx.x * 128;
  const _Float16* q = Q + (size_t)bh * HD * NTOK;
  const _Float16* k = Kg + (size_t)bh * HD * NTOK;
  const _Float16* v = V + (size_t)bh * HD * NTOK;

  const int tid = threadIdx.x;
  const int wid = tid >> 5, lane = tid & 31;
  const int hf = lane >> 4, l15 = lane & 15;

  {  // stage Q transposed: [d][i] -> Qs[i][d]
    const int d = tid >> 2, seg = (tid & 3) * 32;
    const _Float16* src = q + (size_t)d * NTOK + i0 + seg;
#pragma unroll
    for (int j = 0; j < 32; ++j) Qs[seg + j][d] = src[j];
  }
  __syncthreads();

  f16x16 qa[2];
#pragma unroll
  for (int kd = 0; kd < 2; ++kd)
    qa[kd] = ld_afrag(&Qs[wid * 16 + l15][kd * 32], hf);

  float mrow[8], lrow[8];
#pragma unroll
  for (int r = 0; r < 8; ++r) { mrow[r] = -1e30f; lrow[r] = 0.f; }
  f32x8 oacc[4] = {};

  for (int j0 = 0; j0 < NTOK; j0 += 128) {
    __syncthreads();
    {  // stage K tile transposed: [d][j] -> Ks[j][d]
      const int d = tid >> 2, seg = (tid & 3) * 32;
      const _Float16* src = k + (size_t)d * NTOK + j0 + seg;
#pragma unroll
      for (int j = 0; j < 32; ++j) Ks[seg + j][d] = src[j];
    }
    __syncthreads();

    // S[i,j] = sum_d q[d,i] k[d,j] : 8 j-frags x 2 k-steps
    f32x8 s[8];
#pragma unroll
    for (int f = 0; f < 8; ++f) {
      f32x8 c = {};
#pragma unroll
      for (int kd = 0; kd < 2; ++kd) {
        f16x16 bf = ld16(&Ks[f * 16 + l15][kd * 32 + 16 * hf]);
        c = wmma(qa[kd], bf, c);
      }
      s[f] = c;
    }

    // online softmax per row (i = r + 8*half; columns spread over 16 lanes)
#pragma unroll
    for (int r = 0; r < 8; ++r) {
      float mx = -1e30f;
#pragma unroll
      for (int f = 0; f < 8; ++f) mx = fmaxf(mx, s[f][r]);
#pragma unroll
      for (int d = 1; d < 16; d <<= 1) mx = fmaxf(mx, __shfl_xor(mx, d, 32));
      const float mnew = fmaxf(mrow[r], mx);
      const float corr = __expf(mrow[r] - mnew);
      mrow[r] = mnew;
      lrow[r] *= corr;
#pragma unroll
      for (int f = 0; f < 4; ++f) oacc[f][r] = oacc[f][r] * corr;
      float rs = 0.f;
#pragma unroll
      for (int f = 0; f < 8; ++f) {
        const float p = __expf(s[f][r] - mnew);
        s[f][r] = p;
        rs += p;
      }
#pragma unroll
      for (int d = 1; d < 16; d <<= 1) rs += __shfl_xor(rs, d, 32);
      lrow[r] += rs;
    }

    // P -> per-wave LDS (C-layout to A-layout relayout goes through LDS)
#pragma unroll
    for (int f = 0; f < 8; ++f)
#pragma unroll
      for (int r = 0; r < 8; ++r)
        Ps[wid][r + 8 * hf][f * 16 + l15] = (_Float16)s[f][r];
    asm volatile("s_wait_dscnt 0" ::: "memory");  // cross-lane LDS RAW

    // O[i,d] += P[i,j] V[d,j] : A from Ps, B straight from global V
#pragma unroll
    for (int ks = 0; ks < 4; ++ks) {
      f16x16 pa = ld_afrag(&Ps[wid][l15][ks * 32], hf);
#pragma unroll
      for (int fd = 0; fd < 4; ++fd) {
        f16x16 bf = ld16(v + (size_t)(fd * 16 + l15) * NTOK + j0 + ks * 32 + 16 * hf);
        oacc[fd] = wmma(pa, bf, oacc[fd]);
      }
    }
  }

  // write O (f16, head-major [bh][d][i]); D layout: m=i, n=d
#pragma unroll
  for (int fd = 0; fd < 4; ++fd) {
    const int d = fd * 16 + l15;
#pragma unroll
    for (int r = 0; r < 8; ++r) {
      const int i = i0 + wid * 16 + r + 8 * hf;
      O[(size_t)bh * HD * NTOK + (size_t)d * NTOK + i] =
          (_Float16)(oacc[fd][r] / lrow[r]);
    }
  }
}

// ---------------------------------------------------------------------------
extern "C" void kernel_launch(void* const* d_in, const int* in_sizes, int n_in,
                              void* d_out, int out_size, void* d_ws,
                              size_t ws_size, hipStream_t stream) {
  using namespace cfg;
  const float* x      = (const float*)d_in[0];
  const float* norm_w = (const float*)d_in[1];
  const float* norm_b = (const float*)d_in[2];
  const float* qkv_w  = (const float*)d_in[3];
  const float* qkv_b  = (const float*)d_in[4];
  const float* proj_w = (const float*)d_in[5];
  const float* proj_b = (const float*)d_in[6];
  float* out = (float*)d_out;

  // workspace carve-up (f16 activations + f16 weights), 256B aligned
  char* base = (char*)d_ws;
  size_t off = 0;
  auto take = [&](size_t bytes) -> void* {
    void* p = base + off;
    off += (bytes + 255) & ~(size_t)255;
    return p;
  };
  const size_t act = (size_t)B * C * NTOK * sizeof(_Float16);  // 8 MiB each
  _Float16* xn  = (_Float16*)take(act);
  _Float16* qb  = (_Float16*)take(act);
  _Float16* kb  = (_Float16*)take(act);
  _Float16* vb  = (_Float16*)take(act);
  _Float16* ao  = (_Float16*)take(act);
  _Float16* w16 = (_Float16*)take((size_t)3 * C * C * sizeof(_Float16));
  _Float16* p16 = (_Float16*)take((size_t)C * C * sizeof(_Float16));
  (void)ws_size; (void)in_sizes; (void)n_in; (void)out_size;

  // 1) GroupNorm -> f16
  gn_kernel<<<dim3(B * GROUPS), dim3(256), 0, stream>>>(x, norm_w, norm_b, xn);

  // 2) weights -> f16
  cvt_kernel<<<dim3((3 * C * C + 255) / 256), dim3(256), 0, stream>>>(
      qkv_w, proj_w, w16, p16);

  // 3) QKV GEMM (M=1536, N=1024, K=512 per batch)
  gemm_kernel<<<dim3(NTOK / 128, (3 * C) / 128, B), dim3(256), 0, stream>>>(
      w16, xn, qkv_b, qb, kb, vb, nullptr, nullptr, /*mode=*/0);

  // 4) attention (block = 128 queries x one (b,head))
  attn_kernel<<<dim3(NTOK / 128, B * HEADS), dim3(256), 0, stream>>>(
      qb, kb, vb, ao);

  // 5) proj GEMM + bias + residual -> fp32 out
  gemm_kernel<<<dim3(NTOK / 128, C / 128, B), dim3(256), 0, stream>>>(
      p16, ao, proj_b, nullptr, nullptr, nullptr, x, out, /*mode=*/1);
}
